// DynamicGNN_13262859010604
// MI455X (gfx1250) — compile-verified, matrix operand order checked by reading, not run on hardware
//
#include <hip/hip_runtime.h>
#include <hip/hip_bf16.h>

typedef __attribute__((ext_vector_type(16))) _Float16 v16h;
typedef __attribute__((ext_vector_type(8)))  _Float16 v8h;
typedef __attribute__((ext_vector_type(2)))  _Float16 v2h;
typedef __attribute__((ext_vector_type(8)))  float    v8f;

#define BM 128
#define BN 128
#define BK 32
#define LDS_STRIDE 40   // halves; 80 B rows keep 16-B alignment for b128 loads

// ---------------------------------------------------------------------------
// Tiled WMMA GEMM: C[M,Nc] = A[M,K] @ B[K,Nc]  (+bias, +relu optional)
// fp32 in/out, f16 WMMA with fp32 accumulate. 256 threads = 8 wave32s.
// Software-pipelined: double-buffered LDS, next tile staged through registers
// while the current tile's 8 WMMAs execute. A lane's v16h fragment is two
// contiguous 16-byte runs in LDS -> 2x ds_load_b128, no element gather.
// ---------------------------------------------------------------------------
__global__ __launch_bounds__(256)
void gemm_f16_wmma(const float* __restrict__ A, const float* __restrict__ B,
                   const float* __restrict__ bias, float* __restrict__ C,
                   int M, int K, int Nc, int fuse_relu)
{
    __shared__ __align__(16) _Float16 As[2][BM][LDS_STRIDE];
    __shared__ __align__(16) _Float16 Bt[2][BN][LDS_STRIDE];

    const int tid  = threadIdx.x;
    const int wave = tid >> 5;
    const int lane = tid & 31;
    const int rowBase = blockIdx.x * BM;
    const int colBase = blockIdx.y * BN;

    const int nl    = lane & 15;
    const int mloc  = (wave << 4) + nl;      // A-frag row in block tile
    const int khalf = lane >> 4;             // ISA: lanes 16-31 hold K+8 half
    const int kfo   = khalf << 3;            // fragment K offset (halves)

    // staging geometry (per thread, 16 elements each as 8 pairs)
    const int aRow0 = tid >> 4;              // + 16*it
    const int aKp   = (tid & 15) << 1;
    const int bN    = tid & 127;             // fixed column
    const int bKp0  = (tid >> 7) << 1;       // + 4*it

    v8f acc[8];
#pragma unroll
    for (int j = 0; j < 8; ++j) acc[j] = (v8f){0.f,0.f,0.f,0.f,0.f,0.f,0.f,0.f};

    union Frag { v16h v; v8h h[2]; };

    float2 aReg[8];
    float  bReg0[8], bReg1[8];

    auto stageLoad = [&](int k0) {
#pragma unroll
        for (int it = 0; it < 8; ++it) {
            int gr  = rowBase + aRow0 + (it << 4);
            int grc = gr < M ? gr : (M - 1);
            float2 f = *(const float2*)&A[(long long)grc * K + (k0 + aKp)];
            if (gr >= M) { f.x = 0.f; f.y = 0.f; }
            aReg[it] = f;
        }
#pragma unroll
        for (int it = 0; it < 8; ++it) {
            int kp = bKp0 + (it << 2);
            bReg0[it] = B[(long long)(k0 + kp)     * Nc + (colBase + bN)];
            bReg1[it] = B[(long long)(k0 + kp + 1) * Nc + (colBase + bN)];
        }
    };
    auto stageStore = [&](int buf) {
#pragma unroll
        for (int it = 0; it < 8; ++it) {
            v2h pk; pk[0] = (_Float16)aReg[it].x; pk[1] = (_Float16)aReg[it].y;
            *(v2h*)&As[buf][aRow0 + (it << 4)][aKp] = pk;
        }
#pragma unroll
        for (int it = 0; it < 8; ++it) {
            v2h pk; pk[0] = (_Float16)bReg0[it]; pk[1] = (_Float16)bReg1[it];
            *(v2h*)&Bt[buf][bN][bKp0 + (it << 2)] = pk;
        }
    };

    const int iters = K / BK;
    stageLoad(0);
    stageStore(0);
    __syncthreads();

    for (int kt = 0; kt < iters; ++kt) {
        const int cur = kt & 1;
        const bool hasNext = (kt + 1) < iters;
        if (hasNext) stageLoad((kt + 1) * BK);   // global loads overlap WMMAs

        // ---- fragments: contiguous 16-B runs -> ds_load_b128 ----
        Frag af;
        af.h[0] = *(const v8h*)&As[cur][mloc][kfo];
        af.h[1] = *(const v8h*)&As[cur][mloc][kfo + 16];

        Frag bf[8];
#pragma unroll
        for (int j = 0; j < 8; ++j) {
            const int n = (j << 4) + nl;
            bf[j].h[0] = *(const v8h*)&Bt[cur][n][kfo];
            bf[j].h[1] = *(const v8h*)&Bt[cur][n][kfo + 16];
        }
#pragma unroll
        for (int j = 0; j < 8; ++j) {
            acc[j] = __builtin_amdgcn_wmma_f32_16x16x32_f16(
                false, af.v, false, bf[j].v, (short)0, acc[j], false, false);
        }

        if (hasNext) stageStore(cur ^ 1);        // cvt + ds_store next tile
        __syncthreads();
    }

    // epilogue: D layout — element r: m = r + 8*(lane/16), n = lane%16
    const int mh = khalf << 3;
#pragma unroll
    for (int j = 0; j < 8; ++j) {
        int col = colBase + (j << 4) + nl;
        float bv = bias ? bias[col] : 0.f;
#pragma unroll
        for (int r = 0; r < 8; ++r) {
            int row = rowBase + (wave << 4) + r + mh;
            if (row < M) {
                float v = acc[j][r] + bv;
                if (fuse_relu) v = fmaxf(v, 0.f);
                C[(long long)row * Nc + col] = v;
            }
        }
    }
}

// ---------------------------------------------------------------------------
// Small helper kernels
// ---------------------------------------------------------------------------
__global__ void zero_f32(float* __restrict__ p, long long n)
{
    long long i = (long long)blockIdx.x * 256 + threadIdx.x;
    if (i < n) p[i] = 0.f;
}

__global__ void deg_kernel(const int* __restrict__ dst, float* __restrict__ deg, int E)
{
    int e = blockIdx.x * 256 + threadIdx.x;
    if (e < E) atomicAdd(&deg[dst[e]], 1.f);
}

__global__ void dinv_kernel(const float* __restrict__ deg, float* __restrict__ dinv, int N)
{
    int n = blockIdx.x * 256 + threadIdx.x;
    if (n < N) dinv[n] = rsqrtf(deg[n] + 1.f);
}

// 4 edges per 256-thread block; 64 threads/edge, float4 per thread (256 cols)
__global__ __launch_bounds__(256)
void edge_agg(const float* __restrict__ H, const float* __restrict__ dinv,
              const int* __restrict__ src, const int* __restrict__ dst,
              float* __restrict__ agg, int E)
{
    int e = blockIdx.x * 4 + (threadIdx.x >> 6);
    if (e >= E) return;
    int s = src[e];
    int d = dst[e];
    float norm = dinv[s] * dinv[d];
    int c0 = (threadIdx.x & 63) << 2;
    const float4 hv = *(const float4*)(H + (long long)s * 256 + c0);
    float* out = agg + (long long)d * 256 + c0;
    atomicAdd(out + 0, hv.x * norm);
    atomicAdd(out + 1, hv.y * norm);
    atomicAdd(out + 2, hv.z * norm);
    atomicAdd(out + 3, hv.w * norm);
}

// Fused: y = agg + h*dinv^2 + b ; LayerNorm(y)*g+be ; ReLU. One node per block.
__global__ __launch_bounds__(256)
void finalize_ln(const float* __restrict__ agg_in, const float* __restrict__ H,
                 const float* __restrict__ dinv, const float* __restrict__ b,
                 const float* __restrict__ gam, const float* __restrict__ bet,
                 float* __restrict__ Xout)
{
    __shared__ float red[256];
    int n = blockIdx.x;
    int c = threadIdx.x;
    float di = dinv[n];
    float y = agg_in[(long long)n * 256 + c] + H[(long long)n * 256 + c] * di * di + b[c];

    red[c] = y; __syncthreads();
    for (int s = 128; s > 0; s >>= 1) { if (c < s) red[c] += red[c + s]; __syncthreads(); }
    float mean = red[0] * (1.f / 256.f);
    __syncthreads();

    float d0 = y - mean;
    red[c] = d0 * d0; __syncthreads();
    for (int s = 128; s > 0; s >>= 1) { if (c < s) red[c] += red[c + s]; __syncthreads(); }
    float var = red[0] * (1.f / 256.f);

    float v = d0 * rsqrtf(var + 1e-5f) * gam[c] + bet[c];
    Xout[(long long)n * 256 + c] = fmaxf(v, 0.f);
}

// gate[n] = T[n,:128] . Wg2 + bg2   (one wave32 per node)
__global__ __launch_bounds__(256)
void gate_gemv(const float* __restrict__ T, const float* __restrict__ Wg2,
               const float* __restrict__ bg2, float* __restrict__ gate, int N)
{
    int wave = threadIdx.x >> 5, lane = threadIdx.x & 31;
    int n = blockIdx.x * 8 + wave;
    if (n >= N) return;
    const float* t = T + (long long)n * 128;
    float s = 0.f;
#pragma unroll
    for (int p = lane; p < 128; p += 32) s += t[p] * Wg2[p];
#pragma unroll
    for (int off = 16; off > 0; off >>= 1) s += __shfl_xor(s, off, 32);
    if (lane == 0) gate[n] = s + bg2[0];
}

// Per-graph segment softmax + attention pooling (batch is sorted -> binary search)
__global__ __launch_bounds__(256)
void pool_kernel(const float* __restrict__ gate, const int* __restrict__ batch,
                 const float* __restrict__ X, float* __restrict__ pooled, int N)
{
    __shared__ float red[256];
    __shared__ float etile[256];
    __shared__ float s_max, s_inv;
    int g = blockIdx.x, tid = threadIdx.x;

    int lo = 0, hi = N;
    while (lo < hi) { int mid = (lo + hi) >> 1; if (batch[mid] < g) lo = mid + 1; else hi = mid; }
    int start = lo;
    hi = N;
    while (lo < hi) { int mid = (lo + hi) >> 1; if (batch[mid] < g + 1) lo = mid + 1; else hi = mid; }
    int end = lo;

    float mx = -3.402823466e38f;
    for (int n = start + tid; n < end; n += 256) mx = fmaxf(mx, gate[n]);
    red[tid] = mx; __syncthreads();
    for (int s = 128; s > 0; s >>= 1) { if (tid < s) red[tid] = fmaxf(red[tid], red[tid + s]); __syncthreads(); }
    if (tid == 0) s_max = red[0];
    __syncthreads();
    float gm = s_max;

    float ssum = 0.f;
    for (int n = start + tid; n < end; n += 256) ssum += expf(gate[n] - gm);
    red[tid] = ssum; __syncthreads();
    for (int s = 128; s > 0; s >>= 1) { if (tid < s) red[tid] += red[tid + s]; __syncthreads(); }
    if (tid == 0) s_inv = (red[0] > 0.f) ? 1.f / red[0] : 0.f;
    __syncthreads();
    float inv = s_inv;

    float acc = 0.f;
    int c = tid;
    for (int base = start; base < end; base += 256) {
        int cnt = min(256, end - base);
        if (tid < cnt) etile[tid] = expf(gate[base + tid] - gm);
        __syncthreads();
        for (int i = 0; i < cnt; ++i)
            acc += etile[i] * X[(long long)(base + i) * 256 + c];
        __syncthreads();
    }
    pooled[g * 256 + c] = acc * inv;
}

// out[g,o] = C1[g,:128] @ Wc2[:,o] + bc2
__global__ void classifier_out(const float* __restrict__ C1, const float* __restrict__ Wc2,
                               const float* __restrict__ bc2, float* __restrict__ out)
{
    int t = threadIdx.x;
    if (t >= 128) return;
    int g = t >> 1, o = t & 1;
    float s = 0.f;
#pragma unroll 4
    for (int p = 0; p < 128; ++p) s += C1[g * 128 + p] * Wc2[p * 2 + o];
    out[g * 2 + o] = s + bc2[o];
}

// ---------------------------------------------------------------------------
extern "C" void kernel_launch(void* const* d_in, const int* in_sizes, int n_in,
                              void* d_out, int out_size, void* d_ws, size_t ws_size,
                              hipStream_t stream)
{
    const float* x   = (const float*)d_in[0];
    const int*   ei  = (const int*)d_in[1];
    const int*   bat = (const int*)d_in[2];
    const float* W1  = (const float*)d_in[3];
    const float* b1  = (const float*)d_in[4];
    const float* g1  = (const float*)d_in[5];
    const float* be1 = (const float*)d_in[6];
    const float* W2  = (const float*)d_in[7];
    const float* b2  = (const float*)d_in[8];
    const float* g2  = (const float*)d_in[9];
    const float* be2 = (const float*)d_in[10];
    const float* Wg1 = (const float*)d_in[11];
    const float* bg1 = (const float*)d_in[12];
    const float* Wg2 = (const float*)d_in[13];
    const float* bg2 = (const float*)d_in[14];
    const float* Wc1 = (const float*)d_in[15];
    const float* bc1 = (const float*)d_in[16];
    const float* Wc2 = (const float*)d_in[17];
    const float* bc2 = (const float*)d_in[18];

    const int N = 50000, E = 800000, G = 64, F_IN = 128, H = 256, P = 128;
    const int* src = ei;
    const int* dst = ei + E;

    char* ws = (char*)d_ws;
    size_t off = 0;
    auto alloc = [&](size_t bytes) -> void* {
        void* p = ws + off;
        off += (bytes + 255) & ~(size_t)255;
        return p;
    };
    float* bufA   = (float*)alloc((size_t)N * H * 4);  // h (GEMM out), later gate-hidden
    float* bufC   = (float*)alloc((size_t)N * H * 4);  // agg / x (in-place)
    float* dinv   = (float*)alloc((size_t)N * 4);
    float* gateb  = (float*)alloc((size_t)N * 4);      // deg, then gate scores
    float* pooled = (float*)alloc((size_t)G * H * 4);
    float* c1     = (float*)alloc((size_t)G * P * 4);
    if (off > ws_size) return;  // workspace too small: bail deterministically

    float* out = (float*)d_out;
    dim3 blk(256);

    // degree -> dinv
    zero_f32<<<dim3((N + 255) / 256), blk, 0, stream>>>(gateb, N);
    deg_kernel<<<dim3((E + 255) / 256), blk, 0, stream>>>(dst, gateb, E);
    dinv_kernel<<<dim3((N + 255) / 256), blk, 0, stream>>>(gateb, dinv, N);

    // ---- layer 1 ----
    gemm_f16_wmma<<<dim3((N + BM - 1) / BM, H / BN), blk, 0, stream>>>(
        x, W1, nullptr, bufA, N, F_IN, H, 0);
    zero_f32<<<dim3((int)(((long long)N * H + 255) / 256)), blk, 0, stream>>>(bufC, (long long)N * H);
    edge_agg<<<dim3((E + 3) / 4), blk, 0, stream>>>(bufA, dinv, src, dst, bufC, E);
    finalize_ln<<<dim3(N), blk, 0, stream>>>(bufC, bufA, dinv, b1, g1, be1, bufC);

    // ---- layer 2 ----
    gemm_f16_wmma<<<dim3((N + BM - 1) / BM, H / BN), blk, 0, stream>>>(
        bufC, W2, nullptr, bufA, N, H, H, 0);
    zero_f32<<<dim3((int)(((long long)N * H + 255) / 256)), blk, 0, stream>>>(bufC, (long long)N * H);
    edge_agg<<<dim3((E + 3) / 4), blk, 0, stream>>>(bufA, dinv, src, dst, bufC, E);
    finalize_ln<<<dim3(N), blk, 0, stream>>>(bufC, bufA, dinv, b2, g2, be2, bufC);

    // ---- gate MLP ----
    gemm_f16_wmma<<<dim3((N + BM - 1) / BM, P / BN), blk, 0, stream>>>(
        bufC, Wg1, bg1, bufA, N, H, P, 1);
    gate_gemv<<<dim3((N + 7) / 8), blk, 0, stream>>>(bufA, Wg2, bg2, gateb, N);

    // ---- segment softmax + attention pooling ----
    pool_kernel<<<dim3(G), blk, 0, stream>>>(gateb, bat, bufC, pooled, N);

    // ---- classifier ----
    gemm_f16_wmma<<<dim3((G + BM - 1) / BM, P / BN), blk, 0, stream>>>(
        pooled, Wc1, bc1, c1, G, H, P, 1);
    classifier_out<<<dim3(1), dim3(128), 0, stream>>>(c1, Wc2, bc2, out);
}
